// AR_Transcriber_76467597738258
// MI455X (gfx1250) — compile-verified, matrix operand order checked by reading, not run on hardware
//
#include <hip/hip_runtime.h>
#include <hip/hip_bf16.h>
#include <math.h>

// ---------------------------------------------------------------------------
// AR transcriber forward for MI455X (gfx1250, wave32).
//   conv1 (direct) -> conv2/conv3 (implicit-im2col bf16 WMMA, pools folded)
//   -> FC (bf16 WMMA GEMM, epilogue scatters into packed LSTM A-fragments)
//   -> persistent 2-layer LSTM kernel: 96 WGs, weight slices resident in LDS,
//      recurrent state kept in packed WMMA-fragment form in global (L2),
//      cell state in registers, 3 grid barriers per step (L2 atomics),
//      posterior tiles aligned to whole pitches so argmax is barrier-free.
// All matrix math: v_wmma_f32_16x16x32_bf16 (D=f32 acc, A/B=bf16).
// ---------------------------------------------------------------------------

typedef __attribute__((ext_vector_type(16))) __bf16 v16bf;
typedef __attribute__((ext_vector_type(8)))  float  v8f;

__device__ __forceinline__ unsigned short f2bf(float f) {
  union { float f; unsigned int u; } v; v.f = f;
  unsigned int u = v.u;
  unsigned int r = u + 0x7FFFu + ((u >> 16) & 1u);           // RNE
  if ((u & 0x7F800000u) == 0x7F800000u) r = u;               // inf/nan
  return (unsigned short)(r >> 16);
}
__device__ __forceinline__ float bf2f(unsigned short h) {
  union { unsigned int u; float f; } v; v.u = ((unsigned int)h) << 16; return v.f;
}
// K index inside a 32-wide k-tile for fragment element j of `lane`
// (CDNA5 ISA 7.12.2, 16-bit A/B layout; same mapping used for A and B packing)
__device__ __forceinline__ int frag_k(int lane, int j) {
  return (lane < 16) ? ((j < 8) ? j : j + 8) : ((j < 8) ? j + 8 : j + 16);
}
// Inverse map: (in-tile k index kk, matrix row m) -> ushort offset lane*16+j
__device__ __forceinline__ int frag_slot(int kk, int m) {
  int grp, j;
  if (kk < 8)       { grp = 0; j = kk; }
  else if (kk < 16) { grp = 1; j = kk - 8; }
  else if (kk < 24) { grp = 0; j = kk - 8; }
  else              { grp = 1; j = kk - 16; }
  return (grp * 16 + m) * 16 + j;
}
__device__ __forceinline__ float sigmoidf_(float x) { return 1.0f / (1.0f + __expf(-x)); }

__device__ __forceinline__ v8f wmma_bf16(v16bf a, v16bf b, v8f c) {
  return __builtin_amdgcn_wmma_f32_16x16x32_bf16(false, a, false, b, (short)0, c, false, false);
}

// ---------------------------------------------------------------------------
// Weight packing: B-fragment layout [ntile][ktile][lane32][j16] (bf16)
// ---------------------------------------------------------------------------

// LSTM weights, per-WG column permutation: WG wg owns h-units [wg*8, wg*8+8);
// its 32 packed cols are (gate, unit), gate=local/8.
// K layout: [0,768) = wih rows 0..767 | [768,768+kembPad) = wih rows 768+e
// (zero padded past kemb) | then whh rows 0..767.
__global__ void k_pack_lstm(unsigned short* __restrict__ dst,
                            const float* __restrict__ wih,
                            const float* __restrict__ whh,
                            int Ka, int kemb, int kembPad, int KT, int total) {
  int idx = blockIdx.x * blockDim.x + threadIdx.x;
  if (idx >= total) return;
  int per_wg = 2 * KT * 512;
  int wg   = idx / per_wg;
  int rem  = idx % per_wg;
  int tile = rem / (KT * 512);
  int r2   = rem % (KT * 512);
  int kt   = r2 / 512;
  int r3   = r2 % 512;
  int ln = r3 >> 4, j = r3 & 15;
  int local = tile * 16 + (ln & 15);
  int gate = local >> 3, unit = local & 7;
  int col = 768 * gate + wg * 8 + unit;
  int k = kt * 32 + frag_k(ln, j);
  float val = 0.f;
  if (k < 768) {
    val = wih[col * Ka + k];
  } else if (k < 768 + kembPad) {
    int e = k - 768;
    if (e < kemb) val = wih[col * Ka + 768 + e];
  } else {
    int kh = k - 768 - kembPad;
    if (kh < 768) val = whh[col * 768 + kh];
  }
  dst[idx] = f2bf(val);
}

// Plain row-major (ncols, K) weight -> packed B fragments.
__global__ void k_pack_plain(unsigned short* __restrict__ dst,
                             const float* __restrict__ src,
                             int ncols, int K, int KT, int total) {
  int idx = blockIdx.x * blockDim.x + threadIdx.x;
  if (idx >= total) return;
  int nt = idx / (KT * 512);
  int r2 = idx % (KT * 512);
  int kt = r2 / 512;
  int r3 = r2 % 512;
  int ln = r3 >> 4, j = r3 & 15;
  int col = nt * 16 + (ln & 15);
  int k = kt * 32 + frag_k(ln, j);
  float val = (col < ncols && k < K) ? src[col * K + k] : 0.f;
  dst[idx] = f2bf(val);
}

// Posterior weights (440 x 768), pitch-aligned tiles: tile nt's 16 cols are
// (pitch = nt*3 + cl/5, state = cl%5) for cl<15; col 15 = zero pad.
// 30 tiles cover pitches 0..89 (88,89 padded) so each tile owns whole pitches.
__global__ void k_pack_post(unsigned short* __restrict__ dst,   // [30][24][512]
                            const float* __restrict__ src,      // (440,768)
                            int total) {
  int idx = blockIdx.x * blockDim.x + threadIdx.x;
  if (idx >= total) return;
  int nt = idx / (24 * 512);
  int r2 = idx % (24 * 512);
  int kt = r2 / 512;
  int r3 = r2 % 512;
  int ln = r3 >> 4, j = r3 & 15;
  int cl = ln & 15;
  int k = kt * 32 + frag_k(ln, j);
  float val = 0.f;
  if (cl < 15) {
    int pitch = nt * 3 + cl / 5, st = cl % 5;
    if (pitch < 88) val = src[(pitch * 5 + st) * 768 + k];
  }
  dst[idx] = f2bf(val);
}

// Zero recurrent packed state (incl. pad rows m=8..15), prev, barrier.
__global__ void k_zero(unsigned short* H1p, unsigned short* H2p, int* prev, int* bar) {
  int i = blockIdx.x * blockDim.x + threadIdx.x;
  if (i < 2 * 24 * 512) { H1p[i] = 0; H2p[i] = 0; }
  if (i < 2 * 8 * 88) prev[i] = 0;
  if (i < 2) bar[i] = 0;
}

// Zero the acoustic fragment bank (pad rows must be zero; FC fills the rest).
__global__ void k_zero_ac(uint4* AcP, int n4) {
  int i = blockIdx.x * blockDim.x + threadIdx.x;
  if (i < n4) AcP[i] = uint4{0, 0, 0, 0};
}

// ---------------------------------------------------------------------------
// conv1: 1->48 channels, 3x3 SAME, direct (tiny FLOPs), fused BN+ReLU, bf16 out
// ---------------------------------------------------------------------------
__global__ void k_conv1(const float* __restrict__ mel,
                        const float* __restrict__ w, const float* __restrict__ b,
                        const float* __restrict__ g, const float* __restrict__ be,
                        const float* __restrict__ m, const float* __restrict__ v,
                        unsigned short* __restrict__ c1out, int img) {
  int idx = blockIdx.x * blockDim.x + threadIdx.x;
  const int N = 48 * 512 * 229;
  if (idx >= N) return;
  int o = idx / (512 * 229);
  int rem = idx % (512 * 229);
  int t = rem / 229, f = rem % 229;
  float acc = 0.f;
#pragma unroll
  for (int dt = 0; dt < 3; ++dt)
#pragma unroll
    for (int df = 0; df < 3; ++df) {
      int tt = t + dt - 1, ff = f + df - 1;
      if (tt >= 0 && tt < 512 && ff >= 0 && ff < 229)
        acc += w[o * 9 + dt * 3 + df] * mel[(img * 512 + tt) * 229 + ff];
    }
  float y = acc + b[o];
  float sc = g[o] * rsqrtf(v[o] + 1e-5f);
  y = (y - m[o]) * sc + be[o];
  c1out[idx] = f2bf(fmaxf(y, 0.f));
}

// ---------------------------------------------------------------------------
// conv2: implicit-im2col WMMA GEMM. M=512*229 rows, N=48 (3 tiles), K=432.
// ---------------------------------------------------------------------------
__global__ void k_conv2(const unsigned short* __restrict__ c1out,   // [48][512][229]
                        const unsigned short* __restrict__ Bp,      // [3][14][512]
                        const float* __restrict__ bias, const float* __restrict__ g,
                        const float* __restrict__ be, const float* __restrict__ m,
                        const float* __restrict__ v,
                        unsigned short* __restrict__ c2out) {       // [48][512][229]
  __shared__ __align__(32) unsigned short sA[14 * 512];
  const int mtile = blockIdx.x;
  const int tid = threadIdx.x, lane = tid & 31, wave = tid >> 5;
  const int M = 512 * 229;
  for (int idx = tid; idx < 14 * 512; idx += blockDim.x) {
    int kt = idx >> 9, r2 = idx & 511, ln = r2 >> 4, j = r2 & 15;
    int row = mtile * 16 + (ln & 15);
    int k = kt * 32 + frag_k(ln, j);
    unsigned short val = 0;
    if (row < M && k < 432) {
      int t = row / 229, f = row % 229;
      int ci = k / 9, rr = k % 9, dt = rr / 3, df = rr % 3;
      int tt = t + dt - 1, ff = f + df - 1;
      if (tt >= 0 && tt < 512 && ff >= 0 && ff < 229)
        val = c1out[(ci * 512 + tt) * 229 + ff];
    }
    sA[idx] = val;
  }
  __syncthreads();
  v8f acc = {};
  const v16bf* gB = reinterpret_cast<const v16bf*>(Bp) + wave * 14 * 32;
  for (int kt = 0; kt < 14; ++kt) {
    v16bf a = *reinterpret_cast<const v16bf*>(&sA[(kt << 9) + (lane << 4)]);
    v16bf bb = gB[kt * 32 + lane];
    acc = wmma_bf16(a, bb, acc);
  }
  int col = wave * 16 + (lane & 15);                 // out channel < 48
  float sc = g[col] * rsqrtf(v[col] + 1e-5f);
  float b0 = bias[col], mm = m[col], bt = be[col];
  int mbase = mtile * 16 + ((lane >> 4) << 3);
#pragma unroll
  for (int r = 0; r < 8; ++r) {
    int row = mbase + r;
    if (row < M) {
      float y = (acc[r] + b0 - mm) * sc + bt;
      int t = row / 229, f = row % 229;
      c2out[(col * 512 + t) * 229 + f] = f2bf(fmaxf(y, 0.f));
    }
  }
}

// ---------------------------------------------------------------------------
// conv3: same pattern, input = maxpool(conv2) folded into the gather.
// M=512*114 rows, N=96 (6 tiles), K=432. block = 192 (6 waves).
// ---------------------------------------------------------------------------
__global__ void k_conv3(const unsigned short* __restrict__ c2out,   // [48][512][229]
                        const unsigned short* __restrict__ Bp,      // [6][14][512]
                        const float* __restrict__ bias, const float* __restrict__ g,
                        const float* __restrict__ be, const float* __restrict__ m,
                        const float* __restrict__ v,
                        unsigned short* __restrict__ c3out) {       // [96][512][114]
  __shared__ __align__(32) unsigned short sA[14 * 512];
  const int mtile = blockIdx.x;
  const int tid = threadIdx.x, lane = tid & 31, wave = tid >> 5;
  const int M = 512 * 114;
  for (int idx = tid; idx < 14 * 512; idx += blockDim.x) {
    int kt = idx >> 9, r2 = idx & 511, ln = r2 >> 4, j = r2 & 15;
    int row = mtile * 16 + (ln & 15);
    int k = kt * 32 + frag_k(ln, j);
    unsigned short val = 0;
    if (row < M && k < 432) {
      int t = row / 114, f2 = row % 114;
      int ci = k / 9, rr = k % 9, dt = rr / 3, df = rr % 3;
      int tt = t + dt - 1, ff = f2 + df - 1;
      if (tt >= 0 && tt < 512 && ff >= 0 && ff < 114) {
        float a0 = bf2f(c2out[(ci * 512 + tt) * 229 + 2 * ff]);
        float a1 = bf2f(c2out[(ci * 512 + tt) * 229 + 2 * ff + 1]);
        val = f2bf(fmaxf(a0, a1));
      }
    }
    sA[idx] = val;
  }
  __syncthreads();
  v8f acc = {};
  const v16bf* gB = reinterpret_cast<const v16bf*>(Bp) + wave * 14 * 32;
  for (int kt = 0; kt < 14; ++kt) {
    v16bf a = *reinterpret_cast<const v16bf*>(&sA[(kt << 9) + (lane << 4)]);
    v16bf bb = gB[kt * 32 + lane];
    acc = wmma_bf16(a, bb, acc);
  }
  int col = wave * 16 + (lane & 15);                 // out channel < 96
  float sc = g[col] * rsqrtf(v[col] + 1e-5f);
  float b0 = bias[col], mm = m[col], bt = be[col];
  int mbase = mtile * 16 + ((lane >> 4) << 3);
#pragma unroll
  for (int r = 0; r < 8; ++r) {
    int row = mbase + r;
    if (row < M) {
      float y = (acc[r] + b0 - mm) * sc + bt;
      int t = row / 114, f2 = row % 114;
      c3out[(col * 512 + t) * 114 + f2] = f2bf(fmaxf(y, 0.f));
    }
  }
}

// ---------------------------------------------------------------------------
// FC: (512 x 5472) @ (5472 x 768) per image; maxpool2 folded into gather.
// Epilogue scatters straight into the packed acoustic A-fragment bank
// AcP[512][24][512] so the LSTM can feed WMMA directly from global/L2.
// ---------------------------------------------------------------------------
__global__ void k_fc(const unsigned short* __restrict__ c3out,     // [96][512][114]
                     const unsigned short* __restrict__ Fp,        // [48][171][512]
                     const float* __restrict__ fc_b,
                     unsigned short* __restrict__ AcP,             // [512][24][512]
                     int img) {
  extern __shared__ unsigned short sAF[];                          // 171*512
  const int mtile = blockIdx.x;                                    // 0..31
  const int tid = threadIdx.x, lane = tid & 31, wave = tid >> 5;
  for (int idx = tid; idx < 171 * 512; idx += blockDim.x) {
    int kt = idx >> 9, r2 = idx & 511, ln = r2 >> 4, j = r2 & 15;
    int t = mtile * 16 + (ln & 15);
    int k = kt * 32 + frag_k(ln, j);
    unsigned short val = 0;
    if (k < 5472) {
      int c = k / 57, f = k % 57;
      float a0 = bf2f(c3out[(c * 512 + t) * 114 + 2 * f]);
      float a1 = bf2f(c3out[(c * 512 + t) * 114 + 2 * f + 1]);
      val = f2bf(fmaxf(a0, a1));
    }
    sAF[idx] = val;
  }
  __syncthreads();
  for (int nt = wave; nt < 48; nt += 8) {
    v8f acc = {};
    const v16bf* gB = reinterpret_cast<const v16bf*>(Fp) + nt * 171 * 32;
    for (int kt = 0; kt < 171; ++kt) {
      v16bf a = *reinterpret_cast<const v16bf*>(&sAF[(kt << 9) + (lane << 4)]);
      v16bf bb = gB[kt * 32 + lane];
      acc = wmma_bf16(a, bb, acc);
    }
    int col = nt * 16 + (lane & 15);                // feature 0..767 = LSTM K idx
    int akt = col >> 5, slot = frag_slot(col & 31, img);
    float b0 = fc_b[col];
    int mbase = mtile * 16 + ((lane >> 4) << 3);
#pragma unroll
    for (int r = 0; r < 8; ++r) {
      int t = mbase + r;                            // time step
      AcP[((size_t)t * 24 + akt) * 512 + slot] = f2bf(acc[r] + b0);
    }
  }
}

// ---------------------------------------------------------------------------
// Persistent LSTM + posterior + argmax. 3 grid barriers per step.
// ---------------------------------------------------------------------------
#define NWG 96

__device__ __forceinline__ void grid_sync(int* bar, int& lgen) {
  __syncthreads();
  lgen++;
  if (threadIdx.x == 0) {
    __threadfence();
    int prevc = __hip_atomic_fetch_add(&bar[0], 1, __ATOMIC_ACQ_REL, __HIP_MEMORY_SCOPE_AGENT);
    if (prevc == NWG - 1) {
      __hip_atomic_store(&bar[0], 0, __ATOMIC_RELAXED, __HIP_MEMORY_SCOPE_AGENT);
      __hip_atomic_fetch_add(&bar[1], 1, __ATOMIC_RELEASE, __HIP_MEMORY_SCOPE_AGENT);
    } else {
      while (__hip_atomic_load(&bar[1], __ATOMIC_ACQUIRE, __HIP_MEMORY_SCOPE_AGENT) < lgen)
        __builtin_amdgcn_s_sleep(2);
    }
    __threadfence();
  }
  __syncthreads();
}

__global__ void __launch_bounds__(64) k_lstm(
    const unsigned short* __restrict__ gW1,      // [96][2][54][512]
    const unsigned short* __restrict__ gW2,      // [96][2][48][512]
    const unsigned short* __restrict__ gP,       // [30][24][512]
    const unsigned short* __restrict__ AcP,      // [512][24][512]
    const float* __restrict__ bih1, const float* __restrict__ bhh1,
    const float* __restrict__ bih2, const float* __restrict__ bhh2,
    const float* __restrict__ post_b, const float* __restrict__ emb_w,
    unsigned short* __restrict__ H1p,            // [2][24][512] packed h1
    unsigned short* __restrict__ H2p,            // [2][24][512] packed h2
    int* __restrict__ prev,                      // [2][8][88]
    int* __restrict__ bar,
    float* __restrict__ out)                     // [8][512][88][5]
{
  const int wg = blockIdx.x;
  const int tid = threadIdx.x;
  const int lane = tid & 31, wave = tid >> 5;

  extern __shared__ unsigned short smem_us[];
  unsigned short* sW1 = smem_us;                    // 2*54*512 = 55296
  unsigned short* sW2 = sW1 + 2 * 54 * 512;         // 2*48*512 = 49152
  unsigned short* sP  = sW2 + 2 * 48 * 512;         // 24*512   = 12288
  unsigned short* sE  = sP + 24 * 512;              // 6*512    = 3072
  float* szbuf = (float*)(sE + 6 * 512);            // 32 cols * 8 rows
  float* sbias = szbuf + 256;                       // 32 L1 | 32 L2 | 16 post | 10 emb
  int*   sPrev = (int*)(sbias + 96);                // 704 staged prev states

  for (int i = tid; i < 2 * 54 * 512; i += blockDim.x) sW1[i] = gW1[wg * (2 * 54 * 512) + i];
  for (int i = tid; i < 2 * 48 * 512; i += blockDim.x) sW2[i] = gW2[wg * (2 * 48 * 512) + i];
  if (wg < 30)
    for (int i = tid; i < 24 * 512; i += blockDim.x) sP[i] = gP[wg * (24 * 512) + i];
  if (tid < 32) {
    int gate = tid >> 3, unit = tid & 7;
    int col = 768 * gate + wg * 8 + unit;
    sbias[tid] = bih1[col] + bhh1[col];
    sbias[32 + tid] = bih2[col] + bhh2[col];
  }
  if (tid < 16) {
    float bv = 0.f;
    if (wg < 30 && tid < 15) {
      int pitch = wg * 3 + tid / 5, st = tid % 5;
      if (pitch < 88) bv = post_b[pitch * 5 + st];
    }
    sbias[64 + tid] = bv;
  }
  if (tid < 10) sbias[80 + tid] = emb_w[tid];       // (5 states x 2 dims)
  __syncthreads();

  float c1s = 0.f, c2s = 0.f;            // cell state, thread (unit=tid/8, row=tid%8)
  const int u = tid >> 3, rb = tid & 7;
  const int hc = wg * 8 + u;             // owned h column
  const int hoff = ((hc >> 5) << 9) + frag_slot(hc & 31, rb);
  int lgen = 0;

  for (int t = 0; t < 512; ++t) {
    const int rp = t & 1, wp = rp ^ 1;

    // ---- prefetch next step's acoustic fragments toward this WGP (L0/WGP$)
    if (t + 1 < 512) {
      const unsigned short* nx = &AcP[(size_t)(t + 1) * 24 * 512];
      for (int pf = tid; pf < 192; pf += blockDim.x)     // 24KB, 128B/line
        __builtin_prefetch(&nx[pf * 64], 0, 0);
    }

    // ---- stage prev-state vector, then embedding A-fragments (k-tiles 24..29)
    for (int i = tid; i < 704; i += blockDim.x) sPrev[i] = prev[rp * 704 + i];
    __syncthreads();
    for (int idx = tid; idx < 6 * 512; idx += blockDim.x) {
      int kt = idx >> 9, r2 = idx & 511, ln = r2 >> 4, j = r2 & 15;
      int mrow = ln & 15;
      int e = kt * 32 + frag_k(ln, j);            // 0..191
      unsigned short val = 0;
      if (mrow < 8 && e < 176) {
        int s = sPrev[mrow * 88 + (e >> 1)];
        val = f2bf(sbias[80 + s * 2 + (e & 1)]);
      }
      sE[idx] = val;
    }
    __syncthreads();

    // ---- layer 1 WMMA: A = [AcP | sE | H1p(old)], all fragment-packed
    {
      v8f acc = {};
      const unsigned short* wb = &sW1[wave * 54 * 512];
      const unsigned short* ga = &AcP[(size_t)t * 24 * 512];
      for (int kt = 0; kt < 24; ++kt) {
        v16bf a = *reinterpret_cast<const v16bf*>(&ga[(kt << 9) + (lane << 4)]);
        v16bf b = *reinterpret_cast<const v16bf*>(&wb[(kt << 9) + (lane << 4)]);
        acc = wmma_bf16(a, b, acc);
      }
      for (int kt = 0; kt < 6; ++kt) {
        v16bf a = *reinterpret_cast<const v16bf*>(&sE[(kt << 9) + (lane << 4)]);
        v16bf b = *reinterpret_cast<const v16bf*>(&wb[((24 + kt) << 9) + (lane << 4)]);
        acc = wmma_bf16(a, b, acc);
      }
      const unsigned short* gh = &H1p[rp * 12288];
      for (int kt = 0; kt < 24; ++kt) {
        v16bf a = *reinterpret_cast<const v16bf*>(&gh[(kt << 9) + (lane << 4)]);
        v16bf b = *reinterpret_cast<const v16bf*>(&wb[((30 + kt) << 9) + (lane << 4)]);
        acc = wmma_bf16(a, b, acc);
      }
      if (lane < 16) {   // lanes 0-15 hold batch rows 0..7
        int lc = wave * 16 + lane;
        float b0 = sbias[lc];
#pragma unroll
        for (int r = 0; r < 8; ++r) szbuf[lc * 8 + r] = acc[r] + b0;
      }
    }
    __syncthreads();

    // ---- layer 1 cell update -> packed h1_new
    {
      float zi = szbuf[(u) * 8 + rb];
      float zf = szbuf[(8 + u) * 8 + rb];
      float zg = szbuf[(16 + u) * 8 + rb];
      float zo = szbuf[(24 + u) * 8 + rb];
      c1s = sigmoidf_(zf) * c1s + sigmoidf_(zi) * tanhf(zg);
      float h = sigmoidf_(zo) * tanhf(c1s);
      H1p[wp * 12288 + hoff] = f2bf(h);
    }
    grid_sync(bar, lgen);

    // ---- layer 2 WMMA: A = [H1p(new) | H2p(old)] direct from global
    {
      v8f acc = {};
      const unsigned short* wb = &sW2[wave * 48 * 512];
      const unsigned short* g1 = &H1p[wp * 12288];
      const unsigned short* g2 = &H2p[rp * 12288];
      for (int kt = 0; kt < 24; ++kt) {
        v16bf a = *reinterpret_cast<const v16bf*>(&g1[(kt << 9) + (lane << 4)]);
        v16bf b = *reinterpret_cast<const v16bf*>(&wb[(kt << 9) + (lane << 4)]);
        acc = wmma_bf16(a, b, acc);
      }
      for (int kt = 0; kt < 24; ++kt) {
        v16bf a = *reinterpret_cast<const v16bf*>(&g2[(kt << 9) + (lane << 4)]);
        v16bf b = *reinterpret_cast<const v16bf*>(&wb[((24 + kt) << 9) + (lane << 4)]);
        acc = wmma_bf16(a, b, acc);
      }
      if (lane < 16) {
        int lc = wave * 16 + lane;
        float b0 = sbias[32 + lc];
#pragma unroll
        for (int r = 0; r < 8; ++r) szbuf[lc * 8 + r] = acc[r] + b0;
      }
    }
    __syncthreads();

    // ---- layer 2 cell update -> packed h2_new
    {
      float zi = szbuf[(u) * 8 + rb];
      float zf = szbuf[(8 + u) * 8 + rb];
      float zg = szbuf[(16 + u) * 8 + rb];
      float zo = szbuf[(24 + u) * 8 + rb];
      c2s = sigmoidf_(zf) * c2s + sigmoidf_(zi) * tanhf(zg);
      float h = sigmoidf_(zo) * tanhf(c2s);
      H2p[wp * 12288 + hoff] = f2bf(h);
    }
    grid_sync(bar, lgen);

    // ---- posterior (wg<30, wave0): 3 whole pitches per tile -> local argmax
    if (wg < 30 && wave == 0) {
      v8f acc = {};
      const unsigned short* g2 = &H2p[wp * 12288];
      for (int kt = 0; kt < 24; ++kt) {
        v16bf a = *reinterpret_cast<const v16bf*>(&g2[(kt << 9) + (lane << 4)]);
        v16bf b = *reinterpret_cast<const v16bf*>(&sP[(kt << 9) + (lane << 4)]);
        acc = wmma_bf16(a, b, acc);
      }
      if (lane < 15) {                       // cols 0..14 = (pitch-local, state)
        int pl = lane / 5, st = lane % 5;
        int pitch = wg * 3 + pl;
        if (pitch < 88) {
          float b0 = sbias[64 + lane];
#pragma unroll
          for (int r = 0; r < 8; ++r) {
            float lv = acc[r] + b0;
            out[((size_t)r * 512 + t) * 440 + pitch * 5 + st] = lv;
            szbuf[lane * 8 + r] = lv;        // stage for in-WG argmax
          }
        }
      }
    }
    __syncthreads();
    if (wg < 30 && tid < 24) {               // (pitch-local, batch) argmax
      int pl = tid >> 3, b = tid & 7;
      int pitch = wg * 3 + pl;
      if (pitch < 88) {
        int best = 0; float bv = szbuf[(pl * 5) * 8 + b];
#pragma unroll
        for (int s2 = 1; s2 < 5; ++s2) {
          float v2 = szbuf[(pl * 5 + s2) * 8 + b];
          if (v2 > bv) { bv = v2; best = s2; }
        }
        prev[wp * 704 + b * 88 + pitch] = best;
      }
    }
    grid_sync(bar, lgen);
  }
}

// ---------------------------------------------------------------------------
// Host side
// ---------------------------------------------------------------------------
extern "C" void kernel_launch(void* const* d_in, const int* in_sizes, int n_in,
                              void* d_out, int out_size, void* d_ws, size_t ws_size,
                              hipStream_t stream) {
  (void)in_sizes; (void)n_in; (void)out_size; (void)ws_size;

  const float* mel  = (const float*)d_in[0];
  const float* c1w  = (const float*)d_in[1];
  const float* c1b  = (const float*)d_in[2];
  const float* bn1g = (const float*)d_in[3];
  const float* bn1b = (const float*)d_in[4];
  const float* bn1m = (const float*)d_in[5];
  const float* bn1v = (const float*)d_in[6];
  const float* c2w  = (const float*)d_in[7];
  const float* c2b  = (const float*)d_in[8];
  const float* bn2g = (const float*)d_in[9];
  const float* bn2b = (const float*)d_in[10];
  const float* bn2m = (const float*)d_in[11];
  const float* bn2v = (const float*)d_in[12];
  const float* c3w  = (const float*)d_in[13];
  const float* c3b  = (const float*)d_in[14];
  const float* bn3g = (const float*)d_in[15];
  const float* bn3b = (const float*)d_in[16];
  const float* bn3m = (const float*)d_in[17];
  const float* bn3v = (const float*)d_in[18];
  const float* fcw  = (const float*)d_in[19];
  const float* fcb  = (const float*)d_in[20];
  const float* wih1 = (const float*)d_in[21];
  const float* whh1 = (const float*)d_in[22];
  const float* bih1 = (const float*)d_in[23];
  const float* bhh1 = (const float*)d_in[24];
  const float* wih2 = (const float*)d_in[25];
  const float* whh2 = (const float*)d_in[26];
  const float* bih2 = (const float*)d_in[27];
  const float* bhh2 = (const float*)d_in[28];
  const float* postw = (const float*)d_in[29];
  const float* postb = (const float*)d_in[30];
  const float* embw  = (const float*)d_in[31];
  float* out = (float*)d_out;

  char* ws = (char*)d_ws;
  size_t off = 0;
  auto take = [&](size_t bytes) -> char* {
    char* p = ws + off;
    off = (off + bytes + 255) & ~(size_t)255;
    return p;
  };
  unsigned short* W1p = (unsigned short*)take((size_t)96 * 2 * 54 * 512 * 2);
  unsigned short* W2p = (unsigned short*)take((size_t)96 * 2 * 48 * 512 * 2);
  unsigned short* Pp  = (unsigned short*)take((size_t)30 * 24 * 512 * 2);
  unsigned short* Fp  = (unsigned short*)take((size_t)48 * 171 * 512 * 2);
  unsigned short* C2p = (unsigned short*)take((size_t)3 * 14 * 512 * 2);
  unsigned short* C3p = (unsigned short*)take((size_t)6 * 14 * 512 * 2);
  unsigned short* AcP = (unsigned short*)take((size_t)512 * 24 * 512 * 2);
  unsigned short* c1o = (unsigned short*)take((size_t)48 * 512 * 229 * 2);
  unsigned short* c2o = (unsigned short*)take((size_t)48 * 512 * 229 * 2);
  unsigned short* c3o = (unsigned short*)take((size_t)96 * 512 * 114 * 2);
  unsigned short* H1p = (unsigned short*)take((size_t)2 * 24 * 512 * 2);
  unsigned short* H2p = (unsigned short*)take((size_t)2 * 24 * 512 * 2);
  int* prevb = (int*)take((size_t)2 * 8 * 88 * 4);
  int* bar   = (int*)take(256);

  // ---- pack weights into WMMA B-fragment layout (bf16)
  {
    int tot = 96 * 2 * 54 * 512;   // layer1: K = [768 ac | 192 emb-pad | 768 h]
    k_pack_lstm<<<(tot + 255) / 256, 256, 0, stream>>>(W1p, wih1, whh1, 944, 176, 192, 54, tot);
  }
  {
    int tot = 96 * 2 * 48 * 512;   // layer2: K = [768 h1 | 768 h2]
    k_pack_lstm<<<(tot + 255) / 256, 256, 0, stream>>>(W2p, wih2, whh2, 768, 0, 0, 48, tot);
  }
  {
    int tot = 30 * 24 * 512;       // posterior: pitch-aligned tiles
    k_pack_post<<<(tot + 255) / 256, 256, 0, stream>>>(Pp, postw, tot);
  }
  {
    int tot = 48 * 171 * 512;
    k_pack_plain<<<(tot + 255) / 256, 256, 0, stream>>>(Fp, fcw, 768, 5472, 171, tot);
  }
  {
    int tot = 3 * 14 * 512;
    k_pack_plain<<<(tot + 255) / 256, 256, 0, stream>>>(C2p, c2w, 48, 432, 14, tot);
  }
  {
    int tot = 6 * 14 * 512;
    k_pack_plain<<<(tot + 255) / 256, 256, 0, stream>>>(C3p, c3w, 96, 432, 14, tot);
  }
  k_zero<<<96, 256, 0, stream>>>(H1p, H2p, prevb, bar);
  {
    int n4 = (512 * 24 * 512) / 8;   // ushorts / 8 per uint4
    k_zero_ac<<<(n4 + 255) / 256, 256, 0, stream>>>((uint4*)AcP, n4);
  }

  // ---- conv stack + FC, per image (stream-serialized; scratch reused)
  for (int b = 0; b < 8; ++b) {
    k_conv1<<<(48 * 512 * 229 + 255) / 256, 256, 0, stream>>>(
        mel, c1w, c1b, bn1g, bn1b, bn1m, bn1v, c1o, b);
    k_conv2<<<7328, 96, 0, stream>>>(c1o, C2p, c2b, bn2g, bn2b, bn2m, bn2v, c2o);
    k_conv3<<<3648, 192, 0, stream>>>(c2o, C3p, c3b, bn3g, bn3b, bn3m, bn3v, c3o);
    k_fc<<<32, 256, 171 * 512 * 2, stream>>>(c3o, Fp, fcb, AcP, b);
  }

  // ---- persistent LSTM: 96 WGs, ~244KB LDS each (weights resident in LDS)
  size_t lstm_lds = (size_t)(2 * 54 * 512 + 2 * 48 * 512 + 24 * 512 + 6 * 512) * 2
                  + (size_t)(256 + 96 + 704) * 4;
  k_lstm<<<NWG, 64, lstm_lds, stream>>>(W1p, W2p, Pp, AcP,
                                        bih1, bhh1, bih2, bhh2, postb, embw,
                                        H1p, H2p, prevb, bar, out);
}
